// GNNAnomalyDetector_67224828117277
// MI455X (gfx1250) — matrix operand order repaired and checked by reading, not compiled, not run on hardware
//
#include <hip/hip_runtime.h>
#include <hip/hip_bf16.h>
#include <stdint.h>

// ---------------------------------------------------------------------------
// GAT anomaly detector for MI455X (gfx1250, wave32, WMMA).
// Memory-bound on adjacency (268 MB @ 23.3 TB/s ~= 11.5us floor); adj is read
// exactly once per layer (mask kept as 1 bit/elem in registers, e recomputed),
// alpha@h runs on v_wmma_f32_16x16x32_bf16 with a sched_barrier-pinned
// double-buffered load/WMMA pipeline. LDS = 69.9KB -> 4 WGs/WGP occupancy.
// ---------------------------------------------------------------------------

typedef __attribute__((ext_vector_type(16))) __bf16 v16bf;
typedef __attribute__((ext_vector_type(8)))  float  v8f;

#define NEG_BIG (-1.0e9f)
#define LN_EPS  (1e-5f)

union V16BF { uint4 q[2]; v16bf v; };

__device__ __forceinline__ float red16_max(float v) {
  v = fmaxf(v, __shfl_xor(v, 1, 16));
  v = fmaxf(v, __shfl_xor(v, 2, 16));
  v = fmaxf(v, __shfl_xor(v, 4, 16));
  v = fmaxf(v, __shfl_xor(v, 8, 16));
  return v;
}
__device__ __forceinline__ float red16_sum(float v) {
  v += __shfl_xor(v, 1, 16);
  v += __shfl_xor(v, 2, 16);
  v += __shfl_xor(v, 4, 16);
  v += __shfl_xor(v, 8, 16);
  return v;
}

// ---------------------------------------------------------------------------
// proj: h = x @ W  (per node), si = h . a[:64], sj = h . a[64:]
// h written as bf16 in WMMA B-operand swizzled layout:
//   B tile is 32(K) x 16(N); lane = n + 16*(k/16); elem-in-lane = k%16.
//   tile id = ((b*32 + k/32)*4 + n/16); tile = 32 lanes * 16 bf16 = 1KB.
// ---------------------------------------------------------------------------
__global__ __launch_bounds__(256)
void gat_proj_kernel(const float* __restrict__ x, const float* __restrict__ W,
                     const float* __restrict__ a, int Fin,
                     __bf16* __restrict__ hswz,
                     float* __restrict__ si, float* __restrict__ sj) {
  extern __shared__ float psm[];
  float* Ws  = psm;            // Fin*64
  float* a_s = psm + Fin * 64; // 128
  const int tid = threadIdx.x;
  for (int idx = tid; idx < Fin * 64; idx += 256) Ws[idx] = W[idx];
  if (tid < 128) a_s[tid] = a[tid];
  __syncthreads();

  const int node = blockIdx.x * 256 + tid;   // 0..32767
  const int b = node >> 10;
  const int i = node & 1023;
  const float* xr = x + (size_t)node * Fin;

  float h[64];
#pragma unroll
  for (int n = 0; n < 64; ++n) h[n] = 0.f;
  for (int f = 0; f < Fin; ++f) {
    const float xf = xr[f];
    const float* wr = Ws + f * 64;
#pragma unroll
    for (int n = 0; n < 64; ++n) h[n] += xf * wr[n];
  }

  float s_i = 0.f, s_j = 0.f;
#pragma unroll
  for (int n = 0; n < 64; ++n) { s_i += h[n] * a_s[n]; s_j += h[n] * a_s[64 + n]; }
  si[node] = s_i;
  sj[node] = s_j;

  const int kt = i >> 5, kk = i & 31;
  const int elem   = kk & 15;
  const int laneHi = (kk >> 4) << 4;
#pragma unroll
  for (int n = 0; n < 64; ++n) {
    const int nt = n >> 4, nl = n & 15;
    const size_t tile = (size_t)((b * 32 + kt) * 4 + nt);
    hswz[tile * 512 + (size_t)(nl + laneHi) * 16 + elem] = (__bf16)h[n];
  }
}

// ---------------------------------------------------------------------------
// gat_attn: one workgroup = (batch b, 16-row tile). 256 threads = 8 waves.
//   P1a: stream adj once; keep mask as 1 bit/elem in regs (u64/thread); rowmax
//   P1b: recompute e from mask+sj, exp(e-m) -> bf16 P tile in LDS; row sum
//        (1/sum folded into the epilogue -- softmax normalization is linear)
//   P2 : WMMA, 8 waves split K=1024 into 128 each; A/B double-buffered with
//        sched_barrier-pinned pipeline so loads overlap the matrix pipe
//   P3 : reduce partials, scale 1/sum, layernorm(+residual), relu, store
// ---------------------------------------------------------------------------
__global__ __launch_bounds__(256)
void gat_attn_kernel(const float* __restrict__ adj,
                     const __bf16* __restrict__ hswz,
                     const float* __restrict__ si_g,
                     const float* __restrict__ sj_g,
                     const float* __restrict__ gamma,
                     const float* __restrict__ beta,
                     const float* __restrict__ resid, int has_resid,
                     float* __restrict__ y_out) {
  extern __shared__ char smem_raw[];
  float*  sj_s   = (float*)smem_raw;                // 1024 f32      @ 0
  float*  si_s   = (float*)(smem_raw + 4096);       // 16
  float*  rmax_s = (float*)(smem_raw + 4160);       // 16
  float*  rinv_s = (float*)(smem_raw + 4224);       // 16
  float*  part   = (float*)(smem_raw + 4352);       // 8*16*64 f32 = 32KB
  __bf16* P_s    = (__bf16*)(smem_raw + 37120);     // 16*1024 bf16 = 32KB
  // total 69888 bytes -> 4 workgroups per WGP (320KB LDS)

  const int tid  = threadIdx.x;
  const int rt   = blockIdx.x;  // 0..63 row tiles
  const int b    = blockIdx.y;  // 0..31 batches
  const int row0 = rt * 16;

  ((float4*)sj_s)[tid] = ((const float4*)(sj_g + (size_t)b * 1024))[tid];
  if (tid < 16) si_s[tid] = si_g[(size_t)b * 1024 + row0 + tid];
  __syncthreads();

  const int r = tid >> 4;   // row within tile, 0..15 (16 threads per row)
  const int c = tid & 15;
  const float si_r = si_s[r];
  const size_t adj_row = ((size_t)b * 1024 + row0 + r) * 1024;

  // ---- pass 1a: adjacency -> register bitmask + rowmax (adj read ONCE) ----
  uint64_t msk = 0;
  float lmax = -3.0e38f;
#pragma unroll 4
  for (int it = 0; it < 16; ++it) {
    const int j0 = it * 64 + c * 4;
    if (it < 15) __builtin_prefetch(adj + adj_row + j0 + 64, 0, 0);  // global_prefetch_b8
    const float4 a4 = *(const float4*)(adj + adj_row + j0);
    const float4 s4 = *(const float4*)(sj_s + j0);
    const unsigned m0 = (a4.x != 0.f), m1 = (a4.y != 0.f);
    const unsigned m2 = (a4.z != 0.f), m3 = (a4.w != 0.f);
    msk |= (uint64_t)(m0 | (m1 << 1) | (m2 << 2) | (m3 << 3)) << (it * 4);
    float t, ev0, ev1, ev2, ev3;
    t = si_r + s4.x; t = (t > 0.f) ? t : 0.2f * t; ev0 = m0 ? t : NEG_BIG;
    t = si_r + s4.y; t = (t > 0.f) ? t : 0.2f * t; ev1 = m1 ? t : NEG_BIG;
    t = si_r + s4.z; t = (t > 0.f) ? t : 0.2f * t; ev2 = m2 ? t : NEG_BIG;
    t = si_r + s4.w; t = (t > 0.f) ? t : 0.2f * t; ev3 = m3 ? t : NEG_BIG;
    lmax = fmaxf(lmax, fmaxf(fmaxf(ev0, ev1), fmaxf(ev2, ev3)));
  }
  lmax = red16_max(lmax);
  if (c == 0) rmax_s[r] = lmax;
  __syncthreads();

  // ---- pass 1b: recompute e from mask, unnormalized exp -> bf16 P; sum ----
  const float m = rmax_s[r];
  float lsum = 0.f;
#pragma unroll 4
  for (int it = 0; it < 16; ++it) {
    const int j0 = it * 64 + c * 4;
    const float4 s4 = *(const float4*)(sj_s + j0);
    const unsigned mb = (unsigned)(msk >> (it * 4)) & 15u;
    float t, ev0, ev1, ev2, ev3;
    t = si_r + s4.x; t = (t > 0.f) ? t : 0.2f * t; ev0 = (mb & 1u) ? t : NEG_BIG;
    t = si_r + s4.y; t = (t > 0.f) ? t : 0.2f * t; ev1 = (mb & 2u) ? t : NEG_BIG;
    t = si_r + s4.z; t = (t > 0.f) ? t : 0.2f * t; ev2 = (mb & 4u) ? t : NEG_BIG;
    t = si_r + s4.w; t = (t > 0.f) ? t : 0.2f * t; ev3 = (mb & 8u) ? t : NEG_BIG;
    const float p0 = __expf(ev0 - m), p1 = __expf(ev1 - m);
    const float p2 = __expf(ev2 - m), p3 = __expf(ev3 - m);
    lsum += (p0 + p1) + (p2 + p3);
    union { __bf16 h[4]; uint2 u; } pk;
    pk.h[0] = (__bf16)p0; pk.h[1] = (__bf16)p1; pk.h[2] = (__bf16)p2; pk.h[3] = (__bf16)p3;
    *(uint2*)(P_s + r * 1024 + j0) = pk.u;
  }
  lsum = red16_sum(lsum);
  if (c == 0) rinv_s[r] = 1.f / lsum;   // >=1 always (max-subtracted)
  __syncthreads();

  // ---- phase 2: WMMA.  wave w covers K in [w*128, w*128+128) ----
  const int lane  = tid & 31;
  const int wave  = tid >> 5;
  const int rowA  = lane & 15;   // A: lanes 0-15 rows M0-15, lanes 16-31 same rows
  const int khalf = lane >> 4;   //    K halves {0-15} vs {16-31}
  v8f acc[4] = {};               // 16x64 output tile per wave

  const uint4* hq = (const uint4*)hswz;  // tile = 64 uint4
  V16BF Bt[2][4];                // B double buffer (global)
  V16BF At[2];                   // A double buffer (LDS)
  {
    const int kt0 = wave * 4;
#pragma unroll
    for (int nt = 0; nt < 4; ++nt) {
      const uint4* bp = hq + (size_t)((b * 32 + kt0) * 4 + nt) * 64 + (size_t)lane * 2;
      Bt[0][nt].q[0] = bp[0];
      Bt[0][nt].q[1] = bp[1];
    }
    const int abase = rowA * 1024 + kt0 * 32 + khalf * 16;
    At[0].q[0] = *(const uint4*)(P_s + abase);
    At[0].q[1] = *(const uint4*)(P_s + abase + 8);
  }
  __builtin_amdgcn_sched_barrier(0);
#pragma unroll
  for (int q = 0; q < 4; ++q) {
    const int cur = q & 1, nxt = cur ^ 1;
    if (q < 3) {  // issue next k-tile's loads BEFORE consuming current tiles
      const int ktn = wave * 4 + q + 1;
#pragma unroll
      for (int nt = 0; nt < 4; ++nt) {
        const uint4* bp = hq + (size_t)((b * 32 + ktn) * 4 + nt) * 64 + (size_t)lane * 2;
        Bt[nxt][nt].q[0] = bp[0];
        Bt[nxt][nt].q[1] = bp[1];
      }
      const int abase = rowA * 1024 + ktn * 32 + khalf * 16;
      At[nxt].q[0] = *(const uint4*)(P_s + abase);
      At[nxt].q[1] = *(const uint4*)(P_s + abase + 8);
    }
    __builtin_amdgcn_sched_barrier(0);  // pin: loads stay above the WMMAs
#pragma unroll
    for (int nt = 0; nt < 4; ++nt) {
      acc[nt] = __builtin_amdgcn_wmma_f32_16x16x32_bf16(
          false, At[cur].v, false, Bt[cur][nt].v, (short)0, acc[nt], false, false);
    }
    __builtin_amdgcn_sched_barrier(0);
  }

  // partials -> LDS
#pragma unroll
  for (int nt = 0; nt < 4; ++nt)
#pragma unroll
    for (int rr = 0; rr < 8; ++rr) {
      const int mm = khalf * 8 + rr;  // C layout: lanes 0-15 -> M=r, 16-31 -> M=8+r
      part[(wave * 16 + mm) * 64 + nt * 16 + (lane & 15)] = acc[nt][rr];
    }
  __syncthreads();

  // ---- phase 3: reduce 8 partials, scale 1/sum, LN (+residual), relu ----
  const int base = tid * 4;
  const int m3 = base >> 6;     // row 0..15 (16 consecutive threads per row)
  const int n0 = base & 63;
  float v4[4] = {0.f, 0.f, 0.f, 0.f};
#pragma unroll
  for (int w = 0; w < 8; ++w) {
    const float* pw = part + (w * 16 + m3) * 64 + n0;
#pragma unroll
    for (int k = 0; k < 4; ++k) v4[k] += pw[k];
  }
  const float rs = rinv_s[m3];
  float s1 = 0.f, s2 = 0.f;
#pragma unroll
  for (int k = 0; k < 4; ++k) { v4[k] *= rs; s1 += v4[k]; s2 += v4[k] * v4[k]; }
  s1 = red16_sum(s1);
  s2 = red16_sum(s2);
  const float mean = s1 * (1.f / 64.f);
  const float var  = s2 * (1.f / 64.f) - mean * mean;
  const float inv  = rsqrtf(var + LN_EPS);
  const size_t orow = ((size_t)b * 1024 + row0 + m3) * 64 + n0;
#pragma unroll
  for (int k = 0; k < 4; ++k) {
    const int n = n0 + k;
    float yv = (v4[k] - mean) * inv * gamma[n] + beta[n];
    if (has_resid) yv += resid[orow + k];
    y_out[orow + k] = fmaxf(yv, 0.f);
  }
}

// ---------------------------------------------------------------------------
// classifier: mean pool over N, 64->64 relu, 64->2
// ---------------------------------------------------------------------------
__global__ __launch_bounds__(64)
void gat_cls_kernel(const float* __restrict__ y,
                    const float* __restrict__ Wc1, const float* __restrict__ bc1,
                    const float* __restrict__ Wc2, const float* __restrict__ bc2,
                    float* __restrict__ out) {
  __shared__ float emb[64];
  __shared__ float hc[64];
  const int b = blockIdx.x;
  const int n = threadIdx.x;
  const float* yb = y + (size_t)b * 1024 * 64;
  float s = 0.f;
  for (int i = 0; i < 1024; ++i) s += yb[i * 64 + n];
  emb[n] = s * (1.f / 1024.f);
  __syncthreads();
  float acc = bc1[n];
  for (int f = 0; f < 64; ++f) acc += emb[f] * Wc1[f * 64 + n];
  hc[n] = fmaxf(acc, 0.f);
  __syncthreads();
  if (n < 2) {
    float l = bc2[n];
    for (int f = 0; f < 64; ++f) l += hc[f] * Wc2[f * 2 + n];
    out[b * 2 + n] = l;
  }
}

// ---------------------------------------------------------------------------
extern "C" void kernel_launch(void* const* d_in, const int* in_sizes, int n_in,
                              void* d_out, int out_size, void* d_ws, size_t ws_size,
                              hipStream_t stream) {
  (void)in_sizes; (void)n_in; (void)out_size; (void)ws_size;
  const float* x   = (const float*)d_in[0];
  const float* adj = (const float*)d_in[1];
  const float* W1  = (const float*)d_in[2];
  const float* a1  = (const float*)d_in[3];
  const float* W2  = (const float*)d_in[4];
  const float* a2  = (const float*)d_in[5];
  const float* g1  = (const float*)d_in[6];
  const float* b1  = (const float*)d_in[7];
  const float* g2  = (const float*)d_in[8];
  const float* b2  = (const float*)d_in[9];
  const float* Wc1 = (const float*)d_in[10];
  const float* bc1 = (const float*)d_in[11];
  const float* Wc2 = (const float*)d_in[12];
  const float* bc2 = (const float*)d_in[13];
  float* out = (float*)d_out;

  char* ws = (char*)d_ws;
  float*  si   = (float*)ws;                                   // 128 KB
  float*  sj   = (float*)(ws + (128 << 10));                   // 128 KB
  __bf16* hswz = (__bf16*)(ws + (256 << 10));                  // 4 MB
  float*  y1   = (float*)(ws + (256 << 10) + (4 << 20));       // 8 MB
  float*  y2   = (float*)(ws + (256 << 10) + (12 << 20));      // 8 MB

  const dim3 gAttn(64, 32);
  const size_t smemAttn  = 69888;                  // sj+stats+part(32K)+P(32K)
  const size_t smemProj1 = (4 * 64 + 128) * sizeof(float);
  const size_t smemProj2 = (64 * 64 + 128) * sizeof(float);

  gat_proj_kernel<<<128, 256, smemProj1, stream>>>(x, W1, a1, 4, hswz, si, sj);
  gat_attn_kernel<<<gAttn, 256, smemAttn, stream>>>(adj, hswz, si, sj, g1, b1,
                                                    y1, 0, y1);
  gat_proj_kernel<<<128, 256, smemProj2, stream>>>(y1, W2, a2, 64, hswz, si, sj);
  gat_attn_kernel<<<gAttn, 256, smemAttn, stream>>>(adj, hswz, si, sj, g2, b2,
                                                    y1, 1, y2);
  gat_cls_kernel<<<32, 64, 0, stream>>>(y2, Wc1, bc1, Wc2, bc2, out);
}
